// MultiQueryAttention_31756988186668
// MI455X (gfx1250) — compile-verified, matrix operand order checked by reading, not run on hardware
//
#include <hip/hip_runtime.h>
#include <hip/hip_bf16.h>

#define S_LEN 2048
#define HID   1024
#define NQ    32
#define NKV   8
#define DH    32
#define KVDIM (NKV * DH)          // 256
#define EPSLN 1e-5f
#define SCALE 0.17677669529663687f // 32^-0.5

typedef __attribute__((ext_vector_type(16))) __bf16 v16bf;
typedef __attribute__((ext_vector_type(8)))  float  v8f;

union Frag16 { v16bf v; uint4 q[2]; unsigned short h[16]; };

// fp32 -> bf16 (round-to-nearest-even)
__device__ __forceinline__ unsigned short f2bf_u16(float f) {
  unsigned u = __builtin_bit_cast(unsigned, f);
  u += 0x7FFFu + ((u >> 16) & 1u);
  return (unsigned short)(u >> 16);
}

// Load a 16x32 bf16 A/B fragment slice for this lane: two contiguous
// 16-byte reads at element offsets klo and khi from the lane's row pointer.
// A-layout (ISA 7.12.2): lanes 0-15 hold K {0..7,16..23}, lanes 16-31 hold
// K {8..15,24..31}; B is loaded symmetrically (lane = column, same K split).
__device__ __forceinline__ v16bf load_frag(const unsigned short* rowp, int klo, int khi) {
  Frag16 f;
  f.q[0] = *(const uint4*)(rowp + klo);
  f.q[1] = *(const uint4*)(rowp + khi);
  return f.v;
}

__device__ __forceinline__ v8f wmma_bf16(v16bf a, v16bf b, v8f c) {
  return __builtin_amdgcn_wmma_f32_16x16x32_bf16(false, a, false, b, (short)0, c,
                                                 false, false);
}

// ---------------------------------------------------------------- convert ---
__global__ void cvt_f32_bf16(const float* __restrict__ in,
                             unsigned short* __restrict__ out, int n) {
  int i = blockIdx.x * blockDim.x + threadIdx.x;
  if (i < n) out[i] = f2bf_u16(in[i]);
}

// ------------------------------------------------------------------- GEMM ---
// C[m,n] = sum_k A[m,k] * W[n,k] + bias[n]
// A: [M,K] bf16 row-major, W: [N,K] bf16 row-major, C: [M,N] fp32.
// One wave per 32x64 macro-tile (2 M-tiles x 4 N-tiles): per K-step 12 16-B
// loads feed 8 WMMAs (1.5 loads/WMMA vs 4 for a naive 16x16 tile), cutting
// L2 traffic ~2.7x. Grid = (M/32, N/64).
__global__ void gemm_bf16_wmma(const unsigned short* __restrict__ A,
                               const unsigned short* __restrict__ W,
                               const float* __restrict__ bias,
                               float* __restrict__ C, int N, int K) {
  const int lane = threadIdx.x;
  const int half = lane >> 4, lr = lane & 15;
  const int mBase = blockIdx.x * 32;
  const int nBase = blockIdx.y * 64;

  const unsigned short* arow0 = A + (size_t)(mBase + lr) * K;
  const unsigned short* arow1 = A + (size_t)(mBase + 16 + lr) * K;
  const unsigned short* wrow0 = W + (size_t)(nBase + lr) * K;
  const unsigned short* wrow1 = W + (size_t)(nBase + 16 + lr) * K;
  const unsigned short* wrow2 = W + (size_t)(nBase + 32 + lr) * K;
  const unsigned short* wrow3 = W + (size_t)(nBase + 48 + lr) * K;

  v8f acc[2][4];
#pragma unroll
  for (int i = 0; i < 2; ++i)
#pragma unroll
    for (int j = 0; j < 4; ++j) acc[i][j] = {};

  for (int kb = 0; kb < K; kb += 32) {
    const int klo = kb + half * 8;
    const int khi = kb + 16 + half * 8;
    v16bf a0 = load_frag(arow0, klo, khi);
    v16bf a1 = load_frag(arow1, klo, khi);
    v16bf b0 = load_frag(wrow0, klo, khi);
    v16bf b1 = load_frag(wrow1, klo, khi);
    v16bf b2 = load_frag(wrow2, klo, khi);
    v16bf b3 = load_frag(wrow3, klo, khi);
    acc[0][0] = wmma_bf16(a0, b0, acc[0][0]);
    acc[0][1] = wmma_bf16(a0, b1, acc[0][1]);
    acc[0][2] = wmma_bf16(a0, b2, acc[0][2]);
    acc[0][3] = wmma_bf16(a0, b3, acc[0][3]);
    acc[1][0] = wmma_bf16(a1, b0, acc[1][0]);
    acc[1][1] = wmma_bf16(a1, b1, acc[1][1]);
    acc[1][2] = wmma_bf16(a1, b2, acc[1][2]);
    acc[1][3] = wmma_bf16(a1, b3, acc[1][3]);
  }

#pragma unroll
  for (int i = 0; i < 2; ++i) {
#pragma unroll
    for (int t = 0; t < 4; ++t) {
      const int n = nBase + 16 * t + lr;
      const float b = bias[n];
#pragma unroll
      for (int r = 0; r < 8; ++r) {
        const int m = mBase + 16 * i + r + half * 8;
        C[(size_t)m * N + n] = acc[i][t][r] + b;
      }
    }
  }
}

// --------------------------------------------------- LayerNorm + reshape ---
// One wave per (s, head); lane = d. q/k: LN over D then bf16 (row-major).
// v: bf16 transposed to [kvh*D+d][s] so PV B-fragments are key-contiguous.
__global__ void ln_qkv(const float* __restrict__ qraw, const float* __restrict__ kraw,
                       const float* __restrict__ vraw,
                       const float* __restrict__ qnw, const float* __restrict__ qnb,
                       const float* __restrict__ knw, const float* __restrict__ knb,
                       unsigned short* __restrict__ qbf, unsigned short* __restrict__ kbf,
                       unsigned short* __restrict__ vt) {
  const int lane = threadIdx.x & 31;
  const int task = blockIdx.x * (blockDim.x >> 5) + (threadIdx.x >> 5);
  const int NT_Q = S_LEN * NQ;
  const int NT_K = S_LEN * NKV;

  if (task < NT_Q) {
    const int s = task / NQ, hh = task % NQ;
    float x = qraw[(size_t)s * HID + hh * DH + lane];
    float m = x;
    for (int o = 1; o < 32; o <<= 1) m += __shfl_xor(m, o, 32);
    m *= (1.0f / DH);
    float d = x - m;
    float v = d * d;
    for (int o = 1; o < 32; o <<= 1) v += __shfl_xor(v, o, 32);
    v *= (1.0f / DH);
    float y = d * rsqrtf(v + EPSLN) * qnw[lane] + qnb[lane];
    qbf[(size_t)s * HID + hh * DH + lane] = f2bf_u16(y);
  } else if (task < NT_Q + NT_K) {
    const int t = task - NT_Q;
    const int s = t / NKV, hh = t % NKV;
    float x = kraw[(size_t)s * KVDIM + hh * DH + lane];
    float m = x;
    for (int o = 1; o < 32; o <<= 1) m += __shfl_xor(m, o, 32);
    m *= (1.0f / DH);
    float d = x - m;
    float v = d * d;
    for (int o = 1; o < 32; o <<= 1) v += __shfl_xor(v, o, 32);
    v *= (1.0f / DH);
    float y = d * rsqrtf(v + EPSLN) * knw[lane] + knb[lane];
    kbf[(size_t)s * KVDIM + hh * DH + lane] = f2bf_u16(y);
  } else if (task < NT_Q + 2 * NT_K) {
    const int t = task - NT_Q - NT_K;
    const int s = t / NKV, hh = t % NKV;
    float x = vraw[(size_t)s * KVDIM + hh * DH + lane];
    vt[(size_t)(hh * DH + lane) * S_LEN + s] = f2bf_u16(x);
  }
}

// -------------------------------------------------------- flash attention ---
// One wave per (16 query rows, head). D == 32 == WMMA K: 2 score WMMAs +
// 2 PV WMMAs per 32-key step. Online softmax in fp32; P re-shaped
// C-layout -> A-fragment via 1 KB of LDS. Mask stream (512 MB total) is the
// DRAM-bound part -> prefetch next iteration's K/V/mask lines.
__global__ void flash_attn_wmma(const unsigned short* __restrict__ qbf,
                                const unsigned short* __restrict__ kbf,
                                const unsigned short* __restrict__ vt,
                                const float* __restrict__ mask,
                                unsigned short* __restrict__ obf) {
  __shared__ __align__(16) unsigned short ldsP[16 * 32];

  const int lane = threadIdx.x;
  const int half = lane >> 4, lr = lane & 15;
  const int m0 = blockIdx.x * 16;
  const int h  = blockIdx.y;
  const int kvh = h >> 2; // repeat_interleave(NQ/NKV = 4)

  // Q fragment: rows m0..m0+15, K = full head dim (32)
  const unsigned short* qrow = qbf + (size_t)(m0 + lr) * HID + h * DH;
  const v16bf qf = load_frag(qrow, half * 8, 16 + half * 8);

  v8f o0 = {}, o1 = {};
  float rm[8], rl[8];
#pragma unroll
  for (int r = 0; r < 8; ++r) { rm[r] = -__builtin_inff(); rl[r] = 0.0f; }

  const int mrow_base = m0 + half * 8; // C-layout: vgpr r -> row r (+8 for hi half)

  for (int kb = 0; kb < S_LEN; kb += 32) {
    // Prefetch next key block's K/V rows and mask lines (global_prefetch_b8)
    const int kn = kb + 32;
    if (kn < S_LEN) {
      __builtin_prefetch(kbf + (size_t)(kn + lr) * KVDIM + kvh * DH, 0, 0);
      __builtin_prefetch(kbf + (size_t)(kn + 16 + lr) * KVDIM + kvh * DH, 0, 0);
      __builtin_prefetch(vt + (size_t)(kvh * DH + lr) * S_LEN + kn, 0, 0);
      __builtin_prefetch(mask + (size_t)(mrow_base + (lane & 7)) * S_LEN + kn, 0, 0);
    }

    // K^T fragments: lane = key column, contiguous in d
    const unsigned short* k0row = kbf + (size_t)(kb + lr) * KVDIM + kvh * DH;
    const unsigned short* k1row = kbf + (size_t)(kb + 16 + lr) * KVDIM + kvh * DH;
    v16bf kf0 = load_frag(k0row, half * 8, 16 + half * 8);
    v16bf kf1 = load_frag(k1row, half * 8, 16 + half * 8);

    v8f z = {};
    v8f s0 = wmma_bf16(qf, kf0, z);
    v8f s1 = wmma_bf16(qf, kf1, z);

    __syncthreads(); // single-wave WG: S_NOP in HW, compiler fence for ldsP WAR

#pragma unroll
    for (int r = 0; r < 8; ++r) {
      const size_t mrow = (size_t)(mrow_base + r) * S_LEN;
      float a0 = s0[r] * SCALE + mask[mrow + kb + lr];
      float a1 = s1[r] * SCALE + mask[mrow + kb + 16 + lr];
      float mx = fmaxf(a0, a1);
      for (int o = 1; o < 16; o <<= 1) mx = fmaxf(mx, __shfl_xor(mx, o, 32));
      float nm = fmaxf(rm[r], mx);
      float p0 = __expf(a0 - nm);
      float p1 = __expf(a1 - nm);
      float sm = p0 + p1;
      for (int o = 1; o < 16; o <<= 1) sm += __shfl_xor(sm, o, 32);
      float alpha = __expf(rm[r] - nm);
      rl[r] = rl[r] * alpha + sm;
      rm[r] = nm;
      o0[r] *= alpha;
      o1[r] *= alpha;
      // P tile (16 rows x 32 keys) to LDS, row-major bf16
      ldsP[(r + half * 8) * 32 + lr]      = f2bf_u16(p0);
      ldsP[(r + half * 8) * 32 + 16 + lr] = f2bf_u16(p1);
    }

    __syncthreads(); // make P stores visible before the transposing gather

    // P as A-fragment: lane's row = lr, K split per A-layout
    Frag16 pfr;
    pfr.q[0] = *(const uint4*)&ldsP[lr * 32 + half * 8];
    pfr.q[1] = *(const uint4*)&ldsP[lr * 32 + 16 + half * 8];
    v16bf pf = pfr.v;

    // V as B-fragments: lane = output column d, keys contiguous in vt
    const unsigned short* v0row = vt + (size_t)(kvh * DH + lr) * S_LEN + kb;
    const unsigned short* v1row = vt + (size_t)(kvh * DH + 16 + lr) * S_LEN + kb;
    v16bf vf0 = load_frag(v0row, half * 8, 16 + half * 8);
    v16bf vf1 = load_frag(v1row, half * 8, 16 + half * 8);

    o0 = wmma_bf16(pf, vf0, o0);
    o1 = wmma_bf16(pf, vf1, o1);
  }

#pragma unroll
  for (int r = 0; r < 8; ++r) {
    const int m = m0 + r + half * 8;
    const float inv = 1.0f / rl[r];
    obf[(size_t)m * HID + h * DH + lr]      = f2bf_u16(o0[r] * inv);
    obf[(size_t)m * HID + h * DH + 16 + lr] = f2bf_u16(o1[r] * inv);
  }
}

// ---------------------------------------------------------------- launch ---
extern "C" void kernel_launch(void* const* d_in, const int* in_sizes, int n_in,
                              void* d_out, int out_size, void* d_ws, size_t ws_size,
                              hipStream_t stream) {
  (void)in_sizes; (void)n_in; (void)out_size; (void)ws_size;

  const float* hs   = (const float*)d_in[0];
  const float* mask = (const float*)d_in[1];
  const float* Wq   = (const float*)d_in[2];
  const float* bq   = (const float*)d_in[3];
  const float* Wk   = (const float*)d_in[4];
  const float* bk   = (const float*)d_in[5];
  const float* Wv   = (const float*)d_in[6];
  const float* bv   = (const float*)d_in[7];
  const float* Wo   = (const float*)d_in[8];
  const float* bo   = (const float*)d_in[9];
  const float* qnw  = (const float*)d_in[10];
  const float* qnb  = (const float*)d_in[11];
  const float* knw  = (const float*)d_in[12];
  const float* knb  = (const float*)d_in[13];
  float* out = (float*)d_out;

  char* w = (char*)d_ws;
  unsigned short* hs_bf = (unsigned short*)w; w += (size_t)S_LEN * HID * 2;
  unsigned short* wq_bf = (unsigned short*)w; w += (size_t)HID * HID * 2;
  unsigned short* wk_bf = (unsigned short*)w; w += (size_t)KVDIM * HID * 2;
  unsigned short* wv_bf = (unsigned short*)w; w += (size_t)KVDIM * HID * 2;
  unsigned short* wo_bf = (unsigned short*)w; w += (size_t)HID * HID * 2;
  float* q_raw          = (float*)w;          w += (size_t)S_LEN * HID * 4;
  float* k_raw          = (float*)w;          w += (size_t)S_LEN * KVDIM * 4;
  float* v_raw          = (float*)w;          w += (size_t)S_LEN * KVDIM * 4;
  unsigned short* q_bf  = (unsigned short*)w; w += (size_t)S_LEN * HID * 2;
  unsigned short* k_bf  = (unsigned short*)w; w += (size_t)S_LEN * KVDIM * 2;
  unsigned short* vt_bf = (unsigned short*)w; w += (size_t)KVDIM * S_LEN * 2;
  unsigned short* ao_bf = (unsigned short*)w; w += (size_t)S_LEN * HID * 2;

  // 1) fp32 -> bf16 conversions
  cvt_f32_bf16<<<(S_LEN * HID + 255) / 256, 256, 0, stream>>>(hs, hs_bf, S_LEN * HID);
  cvt_f32_bf16<<<(HID * HID + 255) / 256, 256, 0, stream>>>(Wq, wq_bf, HID * HID);
  cvt_f32_bf16<<<(KVDIM * HID + 255) / 256, 256, 0, stream>>>(Wk, wk_bf, KVDIM * HID);
  cvt_f32_bf16<<<(KVDIM * HID + 255) / 256, 256, 0, stream>>>(Wv, wv_bf, KVDIM * HID);
  cvt_f32_bf16<<<(HID * HID + 255) / 256, 256, 0, stream>>>(Wo, wo_bf, HID * HID);

  // 2) QKV projections (WMMA bf16, fp32 accumulate + bias), 32x64 per wave
  gemm_bf16_wmma<<<dim3(S_LEN / 32, HID / 64), 32, 0, stream>>>(hs_bf, wq_bf, bq, q_raw, HID, HID);
  gemm_bf16_wmma<<<dim3(S_LEN / 32, KVDIM / 64), 32, 0, stream>>>(hs_bf, wk_bf, bk, k_raw, KVDIM, HID);
  gemm_bf16_wmma<<<dim3(S_LEN / 32, KVDIM / 64), 32, 0, stream>>>(hs_bf, wv_bf, bv, v_raw, KVDIM, HID);

  // 3) per-head LN (q,k) + V transpose, all to bf16
  const int tasks = S_LEN * NQ + 2 * S_LEN * NKV;
  ln_qkv<<<(tasks + 7) / 8, 256, 0, stream>>>(q_raw, k_raw, v_raw, qnw, qnb, knw, knb,
                                              q_bf, k_bf, vt_bf);

  // 4) attention (flash-style, WMMA bf16, fp32 softmax state)
  flash_attn_wmma<<<dim3(S_LEN / 16, NQ), 32, 0, stream>>>(q_bf, k_bf, vt_bf, mask, ao_bf);

  // 5) output projection -> fp32 d_out
  gemm_bf16_wmma<<<dim3(S_LEN / 32, HID / 64), 32, 0, stream>>>(ao_bf, wo_bf, bo, out, HID, HID);
}
